// VisualMamba_12463995093387
// MI455X (gfx1250) — matrix operand order repaired and checked by reading, not compiled, not run on hardware
//
#include <hip/hip_runtime.h>
#include <hip/hip_bf16.h>
#include <math.h>

// ---------------------------------------------------------------------------
// Vision-Mamba (Vim) forward on gfx1250 (MI455X).
// GEMMs -> v_wmma_f32_16x16x32_f16 (f16 in, f32 acc); scan/elementwise -> VALU.
// ---------------------------------------------------------------------------

#define VD   192     // model dim D
#define VE   384     // expanded dim E
#define VN   16      // SSM state N
#define VL   2       // layers
#define VM   197     // sequence length (196 patches + cls)
#define VB   8       // batch
#define TOK  (VB * VM)          // 1576 tokens
#define NP   196
#define PD   768                // 3*16*16 patch dim
#define VK   3                  // conv kernel
#define EN   (VE * VN)          // 6144

typedef _Float16 half_t;
typedef __attribute__((ext_vector_type(16))) _Float16 v16h;
typedef __attribute__((ext_vector_type(8)))  _Float16 v8h;
typedef __attribute__((ext_vector_type(8)))  float    v8f;

// ---------------------------------------------------------------------------
// f32 -> f16 conversion (weights, once per launch)
// ---------------------------------------------------------------------------
__global__ void vim_cvt_f16(const float* __restrict__ in, half_t* __restrict__ out, int n) {
  int i = blockIdx.x * blockDim.x + threadIdx.x;
  if (i < n) out[i] = (half_t)in[i];
}

// ---------------------------------------------------------------------------
// Build f16 patch matrix: (B, 196, 768) from img (B,3,224,224)
// ---------------------------------------------------------------------------
__global__ void vim_patchify(const float* __restrict__ img, half_t* __restrict__ patches) {
  int idx = blockIdx.x * blockDim.x + threadIdx.x;
  const int total = VB * NP * PD;
  if (idx >= total) return;
  int within = idx % PD;           // c*256 + i*16 + j
  int t      = idx / PD;
  int p      = t % NP;
  int b      = t / NP;
  int c  = within >> 8;            // /256
  int r  = within & 255;
  int ii = r >> 4;
  int jj = r & 15;
  int ph = p / 14, pw = p % 14;
  int row = ph * 16 + ii, col = pw * 16 + jj;
  float v = img[(((size_t)b * 3 + c) * 224 + row) * 224 + col];
  patches[idx] = (half_t)v;
}

// ---------------------------------------------------------------------------
// x = concat(cls, xemb) + pos  -> (B, 197, 192) f32
// ---------------------------------------------------------------------------
__global__ void vim_assemble(const float* __restrict__ xemb, const float* __restrict__ cls,
                             const float* __restrict__ pos, float* __restrict__ x) {
  int idx = blockIdx.x * blockDim.x + threadIdx.x;
  const int total = TOK * VD;
  if (idx >= total) return;
  int d = idx % VD;
  int t = idx / VD;
  int m = t % VM;
  int b = t / VM;
  float v = (m == 0) ? cls[d] : xemb[((size_t)b * NP + (m - 1)) * VD + d];
  x[idx] = v + pos[(size_t)m * VD + d];
}

// ---------------------------------------------------------------------------
// LayerNorm over D=192: one wave per row; emits f16 and/or f32.
// ---------------------------------------------------------------------------
__global__ void vim_layernorm(const float* __restrict__ x, const float* __restrict__ g,
                              const float* __restrict__ bta, half_t* __restrict__ out_h,
                              float* __restrict__ out_f) {
  const int row  = blockIdx.x;
  const int lane = threadIdx.x;          // 32 threads
  const float* xr = x + (size_t)row * VD;
  float vals[6];
  float sum = 0.f;
#pragma unroll
  for (int t = 0; t < 6; ++t) { vals[t] = xr[lane + t * 32]; sum += vals[t]; }
#pragma unroll
  for (int o = 16; o > 0; o >>= 1) sum += __shfl_xor(sum, o, 32);
  const float mu = sum * (1.f / VD);
  float vs = 0.f;
#pragma unroll
  for (int t = 0; t < 6; ++t) { float d = vals[t] - mu; vs += d * d; }
#pragma unroll
  for (int o = 16; o > 0; o >>= 1) vs += __shfl_xor(vs, o, 32);
  const float rs = rsqrtf(vs * (1.f / VD) + 1e-5f);
#pragma unroll
  for (int t = 0; t < 6; ++t) {
    int d = lane + t * 32;
    float y = (vals[t] - mu) * rs * g[d] + bta[d];
    if (out_h) out_h[(size_t)row * VD + d] = (half_t)y;
    if (out_f) out_f[(size_t)row * VD + d] = y;
  }
}

// ---------------------------------------------------------------------------
// WMMA GEMM: out[M,N] = A[M,K](f16) * B[K,N](f16) + bias [+ bias2] [+ res]
// One wave per 16x64 output strip. K multiple of 32, N multiple of 64.
// Fragment layouts per CDNA5 ISA 7.12.2. Partial M tiles: row index is
// CLAMPED (not masked) so EXEC stays all-ones through the WMMA loop; the
// epilogue guards stores. No divergence -> no per-iteration zero-fills.
// ---------------------------------------------------------------------------
__global__ void vim_wmma_gemm(const half_t* __restrict__ A, const half_t* __restrict__ Bw,
                              const float* __restrict__ bias, const float* __restrict__ bias2,
                              const float* __restrict__ res, float* __restrict__ out,
                              int Mdim, int Ndim, int Kdim) {
  const int lane  = threadIdx.x & 31;
  const int mTile = blockIdx.x;
  const int nBase = blockIdx.y * 64;
  int row = mTile * 16 + (lane & 15);
  if (row >= Mdim) row = Mdim - 1;           // clamp: keep EXEC all-ones
  const int kOff  = (lane >> 4) << 3;        // 0 or 8
  const half_t* aRow = A + (size_t)row * Kdim + kOff;
  v8f acc0 = {}, acc1 = {}, acc2 = {}, acc3 = {};

  for (int kk = 0; kk < Kdim; kk += 32) {
    // A fragment: 16x32 f16. lane<16: K[0..7],K[16..23]; lane>=16: K[8..15],K[24..31]
    const half_t* ap = aRow + kk;
    v8h lo = *(const v8h*)(ap);
    v8h hi = *(const v8h*)(ap + 16);
    v16h a;
#pragma unroll
    for (int t = 0; t < 8; ++t) { a[t] = lo[t]; a[t + 8] = hi[t]; }
    __builtin_prefetch(ap + 32, 0, 1);       // next K-tile -> global_prefetch_b8
    // B fragments: 32x16 each; lane holds column n = lane&15, K same as A fragment
    const half_t* bp = Bw + (size_t)(kk + kOff) * Ndim + nBase + (lane & 15);
    v16h b0, b1, b2, b3;
#pragma unroll
    for (int t = 0; t < 8; ++t) {
      const half_t* p0 = bp + (size_t)t * Ndim;
      const half_t* p1 = bp + (size_t)(t + 16) * Ndim;
      b0[t] = p0[0];  b0[t + 8] = p1[0];
      b1[t] = p0[16]; b1[t + 8] = p1[16];
      b2[t] = p0[32]; b2[t + 8] = p1[32];
      b3[t] = p0[48]; b3[t + 8] = p1[48];
    }
    acc0 = __builtin_amdgcn_wmma_f32_16x16x32_f16(false, a, false, b0, (short)0, acc0, false, false);
    acc1 = __builtin_amdgcn_wmma_f32_16x16x32_f16(false, a, false, b1, (short)0, acc1, false, false);
    acc2 = __builtin_amdgcn_wmma_f32_16x16x32_f16(false, a, false, b2, (short)0, acc2, false, false);
    acc3 = __builtin_amdgcn_wmma_f32_16x16x32_f16(false, a, false, b3, (short)0, acc3, false, false);
  }

  // Epilogue: C/D layout -> VGPR r, lane l: M = r + 8*(l>=16), N = l&15
  const int mOff = mTile * 16 + ((lane >> 4) << 3);
  const int nCol = lane & 15;
#pragma unroll
  for (int r = 0; r < 8; ++r) {
    int m = mOff + r;
    if (m >= Mdim) continue;
#pragma unroll
    for (int j = 0; j < 4; ++j) {
      float acc = (j == 0) ? acc0[r] : (j == 1) ? acc1[r] : (j == 2) ? acc2[r] : acc3[r];
      int n = nBase + j * 16 + nCol;
      float v = acc + bias[n];
      if (bias2) v += bias2[n];
      if (res)   v += res[(size_t)m * Ndim + n];
      out[(size_t)m * Ndim + n] = v;
    }
  }
}

// ---------------------------------------------------------------------------
// Depthwise conv (K=3, 'same') along sequence, fwd + bwd, fused SiLU -> f16
// fwd: xf[m] = silu(sum_k wf[k]*xp[m-1+k] + bf)
// bwd: xb[m] = silu(sum_k wb[k]*xp[m+1-k] + bb)
// ---------------------------------------------------------------------------
__device__ __forceinline__ float vim_silu(float x) { return x / (1.f + __expf(-x)); }

__global__ void vim_dwconv(const float* __restrict__ xp,
                           const float* __restrict__ cwf, const float* __restrict__ cbf,
                           const float* __restrict__ cwb, const float* __restrict__ cbb,
                           half_t* __restrict__ xf_h, half_t* __restrict__ xb_h) {
  int idx = blockIdx.x * blockDim.x + threadIdx.x;
  const int total = TOK * VE;
  if (idx >= total) return;
  int e = idx % VE;
  int t = idx / VE;
  int m = t % VM;
  float xm  = xp[idx];
  float xmm = (m > 0)      ? xp[idx - VE] : 0.f;
  float xmp = (m < VM - 1) ? xp[idx + VE] : 0.f;
  const float* wf = cwf + (size_t)e * VK;
  const float* wb = cwb + (size_t)e * VK;
  float yf = wf[0] * xmm + wf[1] * xm + wf[2] * xmp + cbf[e];
  float yb = wb[0] * xmp + wb[1] * xm + wb[2] * xmm + cbb[e];
  xf_h[idx] = (half_t)vim_silu(yf);
  xb_h[idx] = (half_t)vim_silu(yb);
}

// ---------------------------------------------------------------------------
// SSM recurrence (log-space, mirrors reference): one thread per (b,e),
// sequential over m, N=16 state in registers.
// ---------------------------------------------------------------------------
__global__ void vim_scan(const float* __restrict__ Bm, const float* __restrict__ Cm,
                         const float* __restrict__ Dpre, const half_t* __restrict__ xs,
                         const float* __restrict__ Apar, float* __restrict__ yacc,
                         int accumulate) {
  int t = blockIdx.x * blockDim.x + threadIdx.x;
  if (t >= VB * VE) return;
  int e = t % VE;
  int b = t / VE;
  float ap[VN], logP[VN], sgn[VN], S[VN];
#pragma unroll
  for (int n = 0; n < VN; ++n) {
    ap[n] = Apar[(size_t)e * VN + n];
    logP[n] = 0.f; sgn[n] = 1.f; S[n] = 0.f;
  }
  const float eps = 1e-6f;
  for (int m = 0; m < VM; ++m) {
    size_t tok = (size_t)b * VM + m;
    float dp = Dpre[tok * VE + e];
    float Delta = (dp > 20.f) ? dp : log1pf(__expf(dp));   // softplus
    float xv = (float)xs[tok * VE + e];
    const float* bRow = Bm + tok * EN + (size_t)e * VN;
    const float* cRow = Cm + tok * EN + (size_t)e * VN;
    float y = 0.f;
#pragma unroll
    for (int n = 0; n < VN; ++n) {
      float Af = Delta * ap[n];
      float s  = (Af > 0.f) ? 1.f : ((Af < 0.f) ? -1.f : 0.f);
      float absA = fabsf(Af); if (absA < eps) absA = eps;
      logP[n] += __logf(absA);
      sgn[n]  *= s;
      float P    = sgn[n] * __expf(logP[n]);
      float invP = 1.f / (P + eps);
      float Bfull = Delta * bRow[n];
      S[n] += invP * Bfull * xv;
      y    += (P * S[n]) * cRow[n];
    }
    float* o = yacc + tok * VE + e;
    *o = accumulate ? (*o + y) : y;
  }
}

// ---------------------------------------------------------------------------
// y = (y_f + y_b) * silu(z)  -> f16 for output GEMM
// ---------------------------------------------------------------------------
__global__ void vim_gate(const float* __restrict__ yacc, const float* __restrict__ z,
                         half_t* __restrict__ y_h) {
  int idx = blockIdx.x * blockDim.x + threadIdx.x;
  if (idx >= TOK * VE) return;
  y_h[idx] = (half_t)(yacc[idx] * vim_silu(z[idx]));
}

// ---------------------------------------------------------------------------
// Host orchestration
// ---------------------------------------------------------------------------
static inline dim3 gemm_grid(int Mdim, int Ndim) {
  return dim3((Mdim + 15) / 16, Ndim / 64, 1);
}

extern "C" void kernel_launch(void* const* d_in, const int* in_sizes, int n_in,
                              void* d_out, int out_size, void* d_ws, size_t ws_size,
                              hipStream_t stream) {
  // ---- inputs (setup_inputs order) ----
  const float* img     = (const float*)d_in[0];
  const float* patch_w = (const float*)d_in[1];
  const float* patch_b = (const float*)d_in[2];
  const float* cls     = (const float*)d_in[3];
  const float* pos     = (const float*)d_in[4];
  const float* ln_g    = (const float*)d_in[5];
  const float* ln_b    = (const float*)d_in[6];
  const float* Wx      = (const float*)d_in[7];
  const float* bx      = (const float*)d_in[8];
  const float* Wz      = (const float*)d_in[9];
  const float* bz      = (const float*)d_in[10];
  const float* cwf     = (const float*)d_in[11];
  const float* cbf     = (const float*)d_in[12];
  const float* cwb     = (const float*)d_in[13];
  const float* cbb     = (const float*)d_in[14];
  const float* WBf     = (const float*)d_in[15];
  const float* bBf     = (const float*)d_in[16];
  const float* WCf     = (const float*)d_in[17];
  const float* bCf     = (const float*)d_in[18];
  const float* WDf     = (const float*)d_in[19];
  const float* bDf     = (const float*)d_in[20];
  const float* Af      = (const float*)d_in[21];
  const float* dbf     = (const float*)d_in[22];
  const float* WBb     = (const float*)d_in[23];
  const float* bBb     = (const float*)d_in[24];
  const float* WCb     = (const float*)d_in[25];
  const float* bCb     = (const float*)d_in[26];
  const float* WDb     = (const float*)d_in[27];
  const float* bDb     = (const float*)d_in[28];
  const float* Ab      = (const float*)d_in[29];
  const float* dbb     = (const float*)d_in[30];
  const float* Wout    = (const float*)d_in[31];
  const float* bout    = (const float*)d_in[32];
  const float* ng      = (const float*)d_in[33];
  const float* nb      = (const float*)d_in[34];
  (void)in_sizes; (void)n_in; (void)out_size; (void)ws_size;

  // ---- workspace carve ----
  size_t off = 0;
  auto carve = [&](size_t bytes) -> void* {
    void* p = (char*)d_ws + off;
    off += (bytes + 255) & ~(size_t)255;
    return p;
  };
  half_t* patches_h = (half_t*)carve((size_t)VB * NP * PD * 2);
  float*  xemb      = (float*) carve((size_t)VB * NP * VD * 4);
  float*  x         = (float*) carve((size_t)TOK * VD * 4);
  half_t* xn_h      = (half_t*)carve((size_t)TOK * VD * 2);
  float*  xp        = (float*) carve((size_t)TOK * VE * 4);
  float*  z         = (float*) carve((size_t)TOK * VE * 4);
  half_t* xf_h      = (half_t*)carve((size_t)TOK * VE * 2);
  half_t* xb_h      = (half_t*)carve((size_t)TOK * VE * 2);
  float*  Bmat      = (float*) carve((size_t)TOK * EN * 4);
  float*  Cmat      = (float*) carve((size_t)TOK * EN * 4);
  float*  Dpre      = (float*) carve((size_t)TOK * VE * 4);
  float*  yacc      = (float*) carve((size_t)TOK * VE * 4);
  half_t* y_h       = (half_t*)carve((size_t)TOK * VE * 2);
  // f16 weight pool
  half_t* pw_h   = (half_t*)carve((size_t)PD * VD * 2);
  half_t* Wx_h   = (half_t*)carve((size_t)VL * VD * VE * 2);
  half_t* Wz_h   = (half_t*)carve((size_t)VL * VD * VE * 2);
  half_t* WBf_h  = (half_t*)carve((size_t)VL * VE * EN * 2);
  half_t* WCf_h  = (half_t*)carve((size_t)VL * VE * EN * 2);
  half_t* WBb_h  = (half_t*)carve((size_t)VL * VE * EN * 2);
  half_t* WCb_h  = (half_t*)carve((size_t)VL * VE * EN * 2);
  half_t* WDf_h  = (half_t*)carve((size_t)VL * VE * VE * 2);
  half_t* WDb_h  = (half_t*)carve((size_t)VL * VE * VE * 2);
  half_t* Wout_h = (half_t*)carve((size_t)VL * VE * VD * 2);

  auto cvt = [&](const float* src, half_t* dst, size_t n) {
    vim_cvt_f16<<<(unsigned)((n + 255) / 256), 256, 0, stream>>>(src, dst, (int)n);
  };
  // ---- weight conversions ----
  cvt(patch_w, pw_h,   (size_t)PD * VD);
  cvt(Wx,      Wx_h,   (size_t)VL * VD * VE);
  cvt(Wz,      Wz_h,   (size_t)VL * VD * VE);
  cvt(WBf,     WBf_h,  (size_t)VL * VE * EN);
  cvt(WCf,     WCf_h,  (size_t)VL * VE * EN);
  cvt(WBb,     WBb_h,  (size_t)VL * VE * EN);
  cvt(WCb,     WCb_h,  (size_t)VL * VE * EN);
  cvt(WDf,     WDf_h,  (size_t)VL * VE * VE);
  cvt(WDb,     WDb_h,  (size_t)VL * VE * VE);
  cvt(Wout,    Wout_h, (size_t)VL * VE * VD);

  // ---- patch embed ----
  {
    const int n = VB * NP * PD;
    vim_patchify<<<(n + 255) / 256, 256, 0, stream>>>(img, patches_h);
    vim_wmma_gemm<<<gemm_grid(VB * NP, VD), 32, 0, stream>>>(
        patches_h, pw_h, patch_b, nullptr, nullptr, xemb, VB * NP, VD, PD);
    const int na = TOK * VD;
    vim_assemble<<<(na + 255) / 256, 256, 0, stream>>>(xemb, cls, pos, x);
  }

  // ---- layers ----
  const int nTE = TOK * VE;
  for (int i = 0; i < VL; ++i) {
    vim_layernorm<<<TOK, 32, 0, stream>>>(x, ln_g + (size_t)i * VD, ln_b + (size_t)i * VD,
                                          xn_h, nullptr);
    vim_wmma_gemm<<<gemm_grid(TOK, VE), 32, 0, stream>>>(
        xn_h, Wx_h + (size_t)i * VD * VE, bx + (size_t)i * VE, nullptr, nullptr, xp, TOK, VE, VD);
    vim_wmma_gemm<<<gemm_grid(TOK, VE), 32, 0, stream>>>(
        xn_h, Wz_h + (size_t)i * VD * VE, bz + (size_t)i * VE, nullptr, nullptr, z, TOK, VE, VD);
    vim_dwconv<<<(nTE + 255) / 256, 256, 0, stream>>>(
        xp, cwf + (size_t)i * VE * VK, cbf + (size_t)i * VE,
        cwb + (size_t)i * VE * VK, cbb + (size_t)i * VE, xf_h, xb_h);

    // forward branch
    vim_wmma_gemm<<<gemm_grid(TOK, EN), 32, 0, stream>>>(
        xf_h, WBf_h + (size_t)i * VE * EN, bBf + (size_t)i * EN, nullptr, nullptr, Bmat, TOK, EN, VE);
    vim_wmma_gemm<<<gemm_grid(TOK, EN), 32, 0, stream>>>(
        xf_h, WCf_h + (size_t)i * VE * EN, bCf + (size_t)i * EN, nullptr, nullptr, Cmat, TOK, EN, VE);
    vim_wmma_gemm<<<gemm_grid(TOK, VE), 32, 0, stream>>>(
        xf_h, WDf_h + (size_t)i * VE * VE, bDf + (size_t)i * VE, dbf + (size_t)i * VE,
        nullptr, Dpre, TOK, VE, VE);
    vim_scan<<<(VB * VE + 127) / 128, 128, 0, stream>>>(
        Bmat, Cmat, Dpre, xf_h, Af + (size_t)i * VE * VN, yacc, 0);

    // backward branch
    vim_wmma_gemm<<<gemm_grid(TOK, EN), 32, 0, stream>>>(
        xb_h, WBb_h + (size_t)i * VE * EN, bBb + (size_t)i * EN, nullptr, nullptr, Bmat, TOK, EN, VE);
    vim_wmma_gemm<<<gemm_grid(TOK, EN), 32, 0, stream>>>(
        xb_h, WCb_h + (size_t)i * VE * EN, bCb + (size_t)i * EN, nullptr, nullptr, Cmat, TOK, EN, VE);
    vim_wmma_gemm<<<gemm_grid(TOK, VE), 32, 0, stream>>>(
        xb_h, WDb_h + (size_t)i * VE * VE, bDb + (size_t)i * VE, dbb + (size_t)i * VE,
        nullptr, Dpre, TOK, VE, VE);
    vim_scan<<<(VB * VE + 127) / 128, 128, 0, stream>>>(
        Bmat, Cmat, Dpre, xb_h, Ab + (size_t)i * VE * VN, yacc, 1);

    vim_gate<<<(nTE + 255) / 256, 256, 0, stream>>>(yacc, z, y_h);
    // out-proj + residual (reads x as residual, writes x; same-thread RMW is safe)
    vim_wmma_gemm<<<gemm_grid(TOK, VD), 32, 0, stream>>>(
        y_h, Wout_h + (size_t)i * VE * VD, bout + (size_t)i * VD, nullptr, x, x, TOK, VD, VE);
  }

  // ---- final layernorm -> d_out (f32) ----
  vim_layernorm<<<TOK, 32, 0, stream>>>(x, ng, nb, nullptr, (float*)d_out);
}
